// QStack_25761213841785
// MI455X (gfx1250) — matrix operand order repaired and checked by reading, not compiled, not run on hardware
//
#include <hip/hip_runtime.h>
#include <hip/hip_bf16.h>

// ---------------------------------------------------------------------------
// Types for CDNA5 WMMA (wave32): A/B = 16 bf16 per lane, C/D = 8 f32 per lane
// ---------------------------------------------------------------------------
typedef __attribute__((ext_vector_type(16))) __bf16          bf16x16;
typedef __attribute__((ext_vector_type(8)))  float           v8f;
typedef __attribute__((ext_vector_type(8)))  unsigned short  ushortx8;

union ABfrag { bf16x16 v; ushortx8 h[2]; };

__device__ __forceinline__ unsigned short f2bf(float f) {
    unsigned int u = __float_as_uint(f);
    u += 0x7FFFu + ((u >> 16) & 1u);          // round-to-nearest-even
    return (unsigned short)(u >> 16);
}
__device__ __forceinline__ float ldf(float v) { return v; }
__device__ __forceinline__ float ldf(unsigned short h) {
    return __uint_as_float(((unsigned int)h) << 16);
}

// ---------------------------------------------------------------------------
// Implicit-GEMM conv / transposed-conv with bf16 WMMA, fully templated dims so
// all index math folds to shifts / constant-muls (no runtime int division).
//   MODE 0: forward conv, OIHW weights
//   MODE 1: stride-2 transposed conv, IOHW weights (gather formulation)
// GEMM view: M = B*HO*WO, N = CO, K = CI*KH*KW.
// Block: 256 threads = 8 waves; 64x64 output tile, BK=64.
// Wave tile: 16(M) x 32(N) -> 2 accumulators, 4 WMMAs per staging round,
// A-fragment reused across both N-subtiles.
// ---------------------------------------------------------------------------
#define LST 80   // LDS row stride in ushorts: 160B rows, 16B-aligned fragments

template<int MODE, typename TIN, bool RELU,
         int Bn, int CI, int HI, int WI, int CO, int HO, int WO,
         int KH, int KW, int STRIDE, int PAD>
__global__ __launch_bounds__(256)
void conv_wmma(const TIN* __restrict__ in, const float* __restrict__ wgt,
               const float* __restrict__ bias, unsigned short* __restrict__ out)
{
    constexpr int KHW  = KH * KW;
    constexpr int Ktot = CI * KHW;
    constexpr int HOWO = HO * WO;
    constexpr int M    = Bn * HOWO;
    constexpr int BK2  = 64;                  // two 32-wide WMMA K-chunks

    __shared__ unsigned short At[64 * LST];
    __shared__ unsigned short Bt[64 * LST];

    const int tid  = threadIdx.x;
    const int wv   = tid >> 5, lane = tid & 31;
    const int half = lane >> 4, ln = lane & 15;
    const int m0   = blockIdx.x * 64, n0 = blockIdx.y * 64;
    const int msub = wv & 3;                  // 4 M-subtiles of 16
    const int nsub = wv >> 2;                 // 2 N-slabs of 32

    // A loader: each thread fills 16 consecutive k for one m-row
    const int arow = tid >> 2;
    const int ak0  = (tid & 3) * 16;
    const int m    = m0 + arow;
    const int bA   = m / HOWO;                // constant divisor -> shifts
    const int remA = m - bA * HOWO;
    const int ho   = remA / WO, wo = remA - ho * WO;

    // B loader: each thread fills 16 consecutive k for one n-row
    const int brow = tid >> 2;
    const int bk0  = (tid & 3) * 16;
    const int co_b = n0 + brow;

    v8f acc0 = {}, acc1 = {};

    for (int kb = 0; kb < Ktot; kb += BK2) {
        // ---- stage A (im2col patch, bf16) ----
        #pragma unroll
        for (int j = 0; j < 16; j++) {
            int k = kb + ak0 + j;
            float v = 0.f;
            if (k < Ktot) {
                int ci = k / KHW;             // constant divisors
                int r2 = k - ci * KHW;
                int kh = r2 / KW, kw = r2 - kh * KW;
                int ih, iw; bool ok;
                if (MODE == 0) {
                    ih = ho * STRIDE - PAD + kh;
                    iw = wo * STRIDE - PAD + kw;
                    ok = (ih >= 0) & (ih < HI) & (iw >= 0) & (iw < WI);
                } else {
                    int th = ho + PAD - kh, tw = wo + PAD - kw;
                    ih = th >> 1; iw = tw >> 1;
                    ok = (th >= 0) & ((th & 1) == 0) & (ih < HI) &
                         (tw >= 0) & ((tw & 1) == 0) & (iw < WI);
                }
                if (ok) v = ldf(in[((bA * CI + ci) * HI + ih) * WI + iw]);
            }
            At[arow * LST + ak0 + j] = f2bf(v);
        }
        // ---- stage B (weights, bf16, N-major so frag reads are contiguous) --
        #pragma unroll
        for (int j = 0; j < 16; j++) {
            int k = kb + bk0 + j;
            float v = 0.f;
            if (k < Ktot && co_b < CO) {
                int ci = k / KHW;
                int r2 = k - ci * KHW;
                int kh = r2 / KW, kw = r2 - kh * KW;
                v = (MODE == 0) ? wgt[((co_b * CI + ci) * KH + kh) * KW + kw]
                                : wgt[((ci * CO + co_b) * KH + kh) * KW + kw];
            }
            Bt[brow * LST + bk0 + j] = f2bf(v);
        }
        if (kb + BK2 < Ktot)
            __builtin_prefetch(&wgt[co_b * (MODE == 0 ? Ktot : KHW) + kb + BK2], 0, 3);
        __syncthreads();

        // ---- compute: 2 K-chunks x (A x {B0,B1}) = 4 WMMAs ----
        #pragma unroll
        for (int kc = 0; kc < BK2; kc += 32) {
            ABfrag a, b0, b1;
            const unsigned short* ap = &At[(msub * 16 + ln) * LST + kc + half * 8];
            a.h[0] = *(const ushortx8*)ap;           // k = kc + half*8 + 0..7
            a.h[1] = *(const ushortx8*)(ap + 16);    // k = kc + 16 + half*8 + ..
            const unsigned short* bp0 = &Bt[(nsub * 32 + ln) * LST + kc + half * 8];
            b0.h[0] = *(const ushortx8*)bp0;
            b0.h[1] = *(const ushortx8*)(bp0 + 16);
            const unsigned short* bp1 = bp0 + 16 * LST;
            b1.h[0] = *(const ushortx8*)bp1;
            b1.h[1] = *(const ushortx8*)(bp1 + 16);
            acc0 = __builtin_amdgcn_wmma_f32_16x16x32_bf16(
                       false, a.v, false, b0.v, (short)0, acc0, false, false);
            acc1 = __builtin_amdgcn_wmma_f32_16x16x32_bf16(
                       false, a.v, false, b1.v, (short)0, acc1, false, false);
        }
        __syncthreads();
    }

    // ---- epilogue: bias (+ReLU), store bf16 NCHW ----
    const int co0 = n0 + nsub * 32 + ln;
    const int co1 = co0 + 16;
    const float bv0 = bias[co0];
    const float bv1 = bias[co1];
    #pragma unroll
    for (int r = 0; r < 8; r++) {
        int mm = m0 + msub * 16 + half * 8 + r;   // C/D row layout
        if (mm < M) {
            int b2 = mm / HOWO;
            int rr = mm - b2 * HOWO;
            int h2 = rr / WO, w2 = rr - h2 * WO;
            float v0 = acc0[r] + bv0;
            float v1 = acc1[r] + bv1;
            if (RELU) { v0 = v0 > 0.f ? v0 : 0.f; v1 = v1 > 0.f ? v1 : 0.f; }
            out[((b2 * CO + co0) * HO + h2) * WO + w2] = f2bf(v0);
            out[((b2 * CO + co1) * HO + h2) * WO + w2] = f2bf(v1);
        }
    }
}

// ---------------------------------------------------------------------------
// Repack codebooks: embed (c, d, K) f32 -> embedT (c, K, d) bf16, plus |e_k|^2
// ---------------------------------------------------------------------------
__global__ void repack_embed(const float* __restrict__ embed,
                             unsigned short* __restrict__ embedT,
                             float* __restrict__ norms)
{
    int t = blockIdx.x * 256 + threadIdx.x;     // 0..1023 = (c,k)
    if (t >= 1024) return;
    int c = t >> 9, k = t & 511;
    float s = 0.f;
    #pragma unroll 4
    for (int d = 0; d < 128; d++) {
        float v = embed[(c * 128 + d) * 512 + k];
        s += v * v;
        embedT[(c * 512 + k) * 128 + d] = f2bf(v);
    }
    norms[t] = s;
}

// ---------------------------------------------------------------------------
// Fused VQ: dist GEMM (WMMA, 16 points x 512 codes, D=128), argmin, gather.
// grid = (4096 point-tiles, 2 codebooks); block = 256 (8 waves x 64 codes).
// ---------------------------------------------------------------------------
__global__ __launch_bounds__(256)
void vq_kernel(const unsigned short* __restrict__ ze,
               const unsigned short* __restrict__ embedT,
               const float* __restrict__ norms,
               int* __restrict__ idxo, unsigned short* __restrict__ quant)
{
    __shared__ unsigned short At[16 * 136];     // 272B rows (16B-aligned frags)
    __shared__ float sminv[8][16];
    __shared__ int   smini[8][16];
    __shared__ int   sidx[16];

    const int tid = threadIdx.x;
    const int c = blockIdx.y;
    const int p0 = blockIdx.x * 16;

    // stage A: 16 points x 128 dims of z_e (NCHW bf16) into LDS
    {
        int row = tid >> 4, d0 = (tid & 15) * 8;
        int p = p0 + row, b = p >> 10, hw = p & 1023;
        #pragma unroll
        for (int j = 0; j < 8; j++) {
            int d = d0 + j;
            At[row * 136 + d] = ze[(((b * 256) + (c * 128 + d)) << 10) + hw];
        }
    }
    __syncthreads();

    const int wv = tid >> 5, lane = tid & 31, half = lane >> 4, ln = lane & 15;

    ABfrag a[4];
    #pragma unroll
    for (int q = 0; q < 4; q++) {
        const unsigned short* ap = &At[ln * 136 + q * 32 + half * 8];
        a[q].h[0] = *(const ushortx8*)ap;
        a[q].h[1] = *(const ushortx8*)(ap + 16);
    }

    float minv[8]; int mini[8];
    #pragma unroll
    for (int r = 0; r < 8; r++) { minv[r] = 3.4e38f; mini[r] = 0; }

    for (int nt = 0; nt < 4; nt++) {
        int k = wv * 64 + nt * 16 + ln;         // this lane's code column
        v8f acc = {};
        const unsigned short* eb = &embedT[(c * 512 + k) * 128];
        #pragma unroll
        for (int q = 0; q < 4; q++) {
            ABfrag b;
            const unsigned short* bp = eb + q * 32 + half * 8;
            b.h[0] = *(const ushortx8*)bp;
            b.h[1] = *(const ushortx8*)(bp + 16);
            acc = __builtin_amdgcn_wmma_f32_16x16x32_bf16(
                      false, a[q].v, false, b.v, (short)0, acc, false, false);
        }
        float nk = norms[c * 512 + k];
        #pragma unroll
        for (int r = 0; r < 8; r++) {
            float val = nk - 2.f * acc[r];      // |x|^2 constant per point: drop
            if (val < minv[r]) { minv[r] = val; mini[r] = k; }
        }
    }

    // cross-lane min within each 16-lane half (rows 8*half + r)
    #pragma unroll
    for (int mk = 1; mk < 16; mk <<= 1) {
        #pragma unroll
        for (int r = 0; r < 8; r++) {
            float ov = __shfl_xor(minv[r], mk, 32);
            int   oi = __shfl_xor(mini[r], mk, 32);
            if (ov < minv[r] || (ov == minv[r] && oi < mini[r])) {
                minv[r] = ov; mini[r] = oi;
            }
        }
    }
    if (ln == 0) {
        #pragma unroll
        for (int r = 0; r < 8; r++) {
            sminv[wv][half * 8 + r] = minv[r];
            smini[wv][half * 8 + r] = mini[r];
        }
    }
    __syncthreads();

    if (tid < 16) {                              // reduce across 8 waves
        float bv = sminv[0][tid]; int bi = smini[0][tid];
        #pragma unroll
        for (int w = 1; w < 8; w++) {
            float v = sminv[w][tid]; int i = smini[w][tid];
            if (v < bv || (v == bv && i < bi)) { bv = v; bi = i; }
        }
        sidx[tid] = bi;
        idxo[c * 65536 + p0 + tid] = bi;
    }
    __syncthreads();

    // gather quantized vectors -> NCHW bf16 (z_q forward value == quant)
    {
        int row = tid >> 4, d0 = (tid & 15) * 8;
        int p = p0 + row, b = p >> 10, hw = p & 1023;
        const unsigned short* ep = &embedT[(c * 512 + sidx[row]) * 128 + d0];
        #pragma unroll
        for (int j = 0; j < 8; j++)
            quant[(((b * 256) + (c * 128 + d0 + j)) << 10) + hw] = ep[j];
    }
}

// ---------------------------------------------------------------------------
// Final transposed conv 64 -> 3 (tiny N): direct VALU, writes f32 output
// ---------------------------------------------------------------------------
__global__ void dec3_kernel(const unsigned short* __restrict__ in,
                            const float* __restrict__ wgt,
                            const float* __restrict__ bias,
                            float* __restrict__ out, int total)
{
    int t = blockIdx.x * 256 + threadIdx.x;
    if (t >= total) return;
    const int CI = 64, HI = 64, WI = 64, CO = 3, pad = 1;
    int wo = t & 127, ho = (t >> 7) & 127;
    int q = t >> 14, co = q % 3, b = q / 3;
    float s = bias[co];
    for (int kh = 0; kh < 4; kh++) {
        int th = ho + pad - kh;
        if (th < 0 || (th & 1)) continue;
        int ih = th >> 1; if (ih >= HI) continue;
        for (int kw = 0; kw < 4; kw++) {
            int tw = wo + pad - kw;
            if (tw < 0 || (tw & 1)) continue;
            int iw = tw >> 1; if (iw >= WI) continue;
            const unsigned short* ip = &in[(b * CI * HI + ih) * WI + iw];
            const float* wp = &wgt[(co * 4 + kh) * 4 + kw];
            #pragma unroll 4
            for (int ci = 0; ci < CI; ci++)
                s += ldf(ip[ci * HI * WI]) * wp[ci * CO * 16];
        }
    }
    out[t] = s;
}

// ---------------------------------------------------------------------------
// Host-side orchestration (all on `stream`, scratch from d_ws)
// ---------------------------------------------------------------------------
extern "C" void kernel_launch(void* const* d_in, const int* in_sizes, int n_in,
                              void* d_out, int out_size, void* d_ws, size_t ws_size,
                              hipStream_t stream)
{
    const float* x      = (const float*)d_in[0];
    const float* enc_w1 = (const float*)d_in[1];
    const float* enc_b1 = (const float*)d_in[2];
    const float* enc_w2 = (const float*)d_in[3];
    const float* enc_b2 = (const float*)d_in[4];
    const float* enc_w3 = (const float*)d_in[5];
    const float* enc_b3 = (const float*)d_in[6];
    const float* embed  = (const float*)d_in[7];
    const float* dec_w1 = (const float*)d_in[8];
    const float* dec_b1 = (const float*)d_in[9];
    const float* dec_w2 = (const float*)d_in[10];
    const float* dec_b2 = (const float*)d_in[11];
    const float* dec_w3 = (const float*)d_in[12];
    const float* dec_b3 = (const float*)d_in[13];

    char* ws = (char*)d_ws;
    size_t off = 0;
    auto take = [&](size_t bytes) { char* p = ws + off; off = (off + bytes + 255) & ~(size_t)255; return p; };

    unsigned short* h1     = (unsigned short*)take((size_t)64 * 64  * 64 * 64 * 2);  // (64,64,64,64)
    unsigned short* h2     = (unsigned short*)take((size_t)64 * 128 * 32 * 32 * 2);  // (64,128,32,32)
    unsigned short* z_e    = (unsigned short*)take((size_t)64 * 256 * 32 * 32 * 2);  // (64,256,32,32)
    unsigned short* embedT = (unsigned short*)take((size_t)2 * 512 * 128 * 2);
    float*          norms  = (float*)take(1024 * 4);
    int*            idx    = (int*)take((size_t)2 * 65536 * 4);
    unsigned short* quant  = (unsigned short*)take((size_t)64 * 256 * 32 * 32 * 2);
    unsigned short* d1     = (unsigned short*)take((size_t)64 * 128 * 32 * 32 * 2);
    unsigned short* d2     = (unsigned short*)take((size_t)64 * 64  * 64 * 64 * 2);

    // codebook repack + norms
    repack_embed<<<4, 256, 0, stream>>>(embed, embedT, norms);

    // enc1: 3->64, k4 s2 p1, ReLU.  M=262144 -> 4096 m-tiles; N=64 -> 1 n-tile
    conv_wmma<0, float, true, 64, 3, 128, 128, 64, 64, 64, 4, 4, 2, 1>
        <<<dim3(4096, 1), 256, 0, stream>>>(x, enc_w1, enc_b1, h1);

    // enc2: 64->128, k4 s2 p1, ReLU.  M=65536 -> 1024; N=128 -> 2
    conv_wmma<0, unsigned short, true, 64, 64, 64, 64, 128, 32, 32, 4, 4, 2, 1>
        <<<dim3(1024, 2), 256, 0, stream>>>(h1, enc_w2, enc_b2, h2);

    // enc3: 128->256, k3 s1 p1, no ReLU.  N=256 -> 4
    conv_wmma<0, unsigned short, false, 64, 128, 32, 32, 256, 32, 32, 3, 3, 1, 1>
        <<<dim3(1024, 4), 256, 0, stream>>>(h2, enc_w3, enc_b3, z_e);

    // VQ: 65536 points / 16 per tile, 2 codebooks
    vq_kernel<<<dim3(4096, 2), 256, 0, stream>>>(z_e, embedT, norms, idx, quant);

    // dec1: 256->128, k3 s1 p1, ReLU
    conv_wmma<0, unsigned short, true, 64, 256, 32, 32, 128, 32, 32, 3, 3, 1, 1>
        <<<dim3(1024, 2), 256, 0, stream>>>(quant, dec_w1, dec_b1, d1);

    // dec2: transposed 128->64 up2, k4, ReLU.  M=262144 -> 4096; N=64 -> 1
    conv_wmma<1, unsigned short, true, 64, 128, 32, 32, 64, 64, 64, 4, 4, 2, 1>
        <<<dim3(4096, 1), 256, 0, stream>>>(d1, dec_w2, dec_b2, d2);

    // dec3: transposed 64->3 up2, k4, f32 output (direct)
    int total = 64 * 3 * 128 * 128;
    dec3_kernel<<<(total + 255) / 256, 256, 0, stream>>>(
        d2, dec_w3, dec_b3, (float*)d_out, total);
}
